// EquivariantLayer_59631325937738
// MI455X (gfx1250) — compile-verified, matrix operand order checked by reading, not run on hardware
//
#include <hip/hip_runtime.h>
#include <math.h>

// ---------------------------------------------------------------------------
// Equivariant GNN layer for MI455X (gfx1250, wave32).
// Restructured: per-edge weight GEMMs hoisted to per-node XW tables (16x fewer
// FLOPs), node GEMMs via v_wmma_f32_16x16x4_f32, edge kernel is an L2-resident
// gather + tiny CG contraction + f32 atomic scatter. W3J (real CG) tensors are
// recomputed exactly on-device in fp64 each launch.
// ---------------------------------------------------------------------------

#define N_NODES 5000
#define N_EDGES 80000
#define DIM_NODE 480          // 128 + 3*64 + 5*32
#define XW_STRIDE 2368        // floats of XW table per node
#define W3J_TOTAL 363

typedef __attribute__((ext_vector_type(2)))  float    v2f;
typedef __attribute__((ext_vector_type(8)))  float    v8f;
typedef __attribute__((ext_vector_type(16))) _Float16 v16h;

#if __has_builtin(__builtin_amdgcn_wmma_f32_16x16x4_f32)
#define HAVE_WMMA_F32 1
#else
#define HAVE_WMMA_F32 0
#endif

// ------------------------------ W3J init -----------------------------------
__constant__ int PL1[11] = {0,0,0,1,1,1,1,2,2,2,2};
__constant__ int PL2[11] = {0,1,2,0,1,1,2,0,1,2,2};
__constant__ int PL3[11] = {0,1,2,1,0,2,1,2,1,0,2};
__constant__ int POFF[11] = {0,1,10,35,44,53,98,143,168,213,238};

__device__ inline double dfact(int n) {
  const double f[10] = {1.,1.,2.,6.,24.,120.,720.,5040.,40320.,362880.};
  return f[n];
}

__device__ double su2_cg(int j1,int m1,int j2,int m2,int j3,int m3) {
  if (m1 + m2 != m3) return 0.0;
  if (j3 < abs(j1 - j2) || j3 > j1 + j2) return 0.0;
  double pre = sqrt((2.0*j3+1.0)*dfact(j3+j1-j2)*dfact(j3-j1+j2)*dfact(j1+j2-j3)/dfact(j1+j2+j3+1));
  pre *= sqrt(dfact(j3+m3)*dfact(j3-m3)*dfact(j1-m1)*dfact(j1+m1)*dfact(j2-m2)*dfact(j2+m2));
  double s = 0.0;
  for (int k = 0; k <= j1 + j2 - j3; ++k) {
    int t0=k, t1=j1+j2-j3-k, t2=j1-m1-k, t3=j2+m2-k, t4=j3-j2+m1+k, t5=j3-j1-m2+k;
    if (t0<0||t1<0||t2<0||t3<0||t4<0||t5<0) continue;
    double d = dfact(t0)*dfact(t1)*dfact(t2)*dfact(t3)*dfact(t4)*dfact(t5);
    s += ((k & 1) ? -1.0 : 1.0) / d;
  }
  return pre * s;
}

// element (r,c) of the complex->real basis matrix _cbasis(l)
__device__ void cbasis_elem(int l, int r, int c, double &re, double &im) {
  re = 0.0; im = 0.0;
  const double r2 = 0.70710678118654752440;
  if (r == l) { if (c == l) re = 1.0; return; }
  if (r > l) {
    int m = r - l;
    if (c == l + m) re = (m & 1) ? -r2 : r2;
    else if (c == l - m) re = r2;
    return;
  }
  int m = l - r;                 // row l-m
  if (c == l - m) im = r2;                       // A[l-m,l-m] = i/sqrt2
  else if (c == l + m) im = (m & 1) ? r2 : -r2;  // A[l-m,l+m] = -i(-1)^m/sqrt2
}

__global__ __launch_bounds__(128) void k_w3j(float* __restrict__ w3j) {
  __shared__ double sre[W3J_TOTAL];
  __shared__ double sim[W3J_TOTAL];
  const int tid = threadIdx.x;
  for (int t = tid; t < W3J_TOTAL; t += 128) {
    int p = 10;
    while (t < POFF[p]) --p;
    const int l1 = PL1[p], l2 = PL2[p], l3 = PL3[p];
    const int n2 = 2*l2+1, n3 = 2*l3+1;
    const int rel = t - POFF[p];
    const int c = rel % n3;
    const int b = (rel / n3) % n2;
    const int a = rel / (n3 * n2);
    double sr = 0.0, si = 0.0;
    for (int M = 0; M <= 2*l1; ++M)
      for (int N = 0; N <= 2*l2; ++N)
        for (int P = 0; P <= 2*l3; ++P) {
          double cg = su2_cg(l1, M-l1, l2, N-l2, l3, P-l3);
          if (cg == 0.0) continue;
          double a1r,a1i,a2r,a2i,a3r,a3i;
          cbasis_elem(l1, a, M, a1r, a1i);
          cbasis_elem(l2, b, N, a2r, a2i);
          cbasis_elem(l3, c, P, a3r, a3i);
          a3i = -a3i; // conj
          double tr = a1r*a2r - a1i*a2i;
          double ti = a1r*a2i + a1i*a2r;
          double ur = tr*a3r - ti*a3i;
          double ui = tr*a3i + ti*a3r;
          sr += ur * cg; si += ui * cg;
        }
    sre[t] = sr; sim[t] = si;
  }
  __syncthreads();
  if (tid < 11) {
    const int p = tid;
    const int sz = (2*PL1[p]+1) * (2*PL2[p]+1) * (2*PL3[p]+1);
    const int o = POFF[p];
    double sar = 0.0, sai = 0.0;
    for (int i = 0; i < sz; ++i) { sar += fabs(sre[o+i]); sai += fabs(sim[o+i]); }
    const bool useRe = (sar >= sai);
    double nrm = 0.0;
    for (int i = 0; i < sz; ++i) { double v = useRe ? sre[o+i] : sim[o+i]; nrm += v*v; }
    nrm = sqrt(nrm);
    for (int i = 0; i < sz; ++i) {
      double v = useRe ? sre[o+i] : sim[o+i];
      w3j[o+i] = (float)(v / nrm);
    }
  }
}

// ------------------------------ WMMA GEMMs ---------------------------------
// Computes XW_p(n,w,i) = scale * sum_u Xview(n,u) * W(u,w) for all 11 paths /
// i-components, plus the Wself o3_linear (seeding h before edge scatter-add).
struct GDesc {
  int x_off, x_stride, U, Wd, widx;
  int dst_off, dst_stride, dst_row_stride, dst_buf;  // dst_buf: 0->xw, 1->h
  float scale;
};
struct WPtrs { const float* p[14]; };

__constant__ GDesc g_desc[44] = {
  {0,1,128,128, 0,    0,1,XW_STRIDE,0, 0.05103104f},
  {0,1,128, 64, 1,  128,1,XW_STRIDE,0, 0.04419417f},
  {0,1,128, 32, 2,  192,1,XW_STRIDE,0, 0.04419417f},
  {128,3,64,64, 3,  224,3,XW_STRIDE,0, 0.0625f},
  {129,3,64,64, 3,  225,3,XW_STRIDE,0, 0.0625f},
  {130,3,64,64, 3,  226,3,XW_STRIDE,0, 0.0625f},
  {128,3,64,128,4,  416,3,XW_STRIDE,0, 0.07216878f},
  {129,3,64,128,4,  417,3,XW_STRIDE,0, 0.07216878f},
  {130,3,64,128,4,  418,3,XW_STRIDE,0, 0.07216878f},
  {128,3,64,32, 5,  800,3,XW_STRIDE,0, 0.0625f},
  {129,3,64,32, 5,  801,3,XW_STRIDE,0, 0.0625f},
  {130,3,64,32, 5,  802,3,XW_STRIDE,0, 0.0625f},
  {128,3,64,64, 6,  896,3,XW_STRIDE,0, 0.0625f},
  {129,3,64,64, 6,  897,3,XW_STRIDE,0, 0.0625f},
  {130,3,64,64, 6,  898,3,XW_STRIDE,0, 0.0625f},
  {320,5,32,32, 7, 1088,5,XW_STRIDE,0, 0.08838835f},
  {321,5,32,32, 7, 1089,5,XW_STRIDE,0, 0.08838835f},
  {322,5,32,32, 7, 1090,5,XW_STRIDE,0, 0.08838835f},
  {323,5,32,32, 7, 1091,5,XW_STRIDE,0, 0.08838835f},
  {324,5,32,32, 7, 1092,5,XW_STRIDE,0, 0.08838835f},
  {320,5,32,64, 8, 1248,5,XW_STRIDE,0, 0.08838835f},
  {321,5,32,64, 8, 1249,5,XW_STRIDE,0, 0.08838835f},
  {322,5,32,64, 8, 1250,5,XW_STRIDE,0, 0.08838835f},
  {323,5,32,64, 8, 1251,5,XW_STRIDE,0, 0.08838835f},
  {324,5,32,64, 8, 1252,5,XW_STRIDE,0, 0.08838835f},
  {320,5,32,128,9, 1568,5,XW_STRIDE,0, 0.10206207f},
  {321,5,32,128,9, 1569,5,XW_STRIDE,0, 0.10206207f},
  {322,5,32,128,9, 1570,5,XW_STRIDE,0, 0.10206207f},
  {323,5,32,128,9, 1571,5,XW_STRIDE,0, 0.10206207f},
  {324,5,32,128,9, 1572,5,XW_STRIDE,0, 0.10206207f},
  {320,5,32,32,10, 2208,5,XW_STRIDE,0, 0.08838835f},
  {321,5,32,32,10, 2209,5,XW_STRIDE,0, 0.08838835f},
  {322,5,32,32,10, 2210,5,XW_STRIDE,0, 0.08838835f},
  {323,5,32,32,10, 2211,5,XW_STRIDE,0, 0.08838835f},
  {324,5,32,32,10, 2212,5,XW_STRIDE,0, 0.08838835f},
  // Wself -> h  (h is then accumulated into by the edge kernel)
  {0,1,128,128,11,   0,1,DIM_NODE,1, 0.08838835f},
  {128,3,64,64,12, 128,3,DIM_NODE,1, 0.125f},
  {129,3,64,64,12, 129,3,DIM_NODE,1, 0.125f},
  {130,3,64,64,12, 130,3,DIM_NODE,1, 0.125f},
  {320,5,32,32,13, 320,5,DIM_NODE,1, 0.17677670f},
  {321,5,32,32,13, 321,5,DIM_NODE,1, 0.17677670f},
  {322,5,32,32,13, 322,5,DIM_NODE,1, 0.17677670f},
  {323,5,32,32,13, 323,5,DIM_NODE,1, 0.17677670f},
  {324,5,32,32,13, 324,5,DIM_NODE,1, 0.17677670f},
};

__global__ __launch_bounds__(32) void k_gemm(const float* __restrict__ x, WPtrs wp,
                                             float* __restrict__ xw, float* __restrict__ h) {
  const GDesc d = g_desc[blockIdx.y];
  const float* __restrict__ W = wp.p[d.widx];
  float* __restrict__ dst = d.dst_buf ? h : xw;
  const int lane = threadIdx.x & 31;
  const int half = lane >> 4;      // wave32: lanes 0-15 / 16-31 hold K halves
  const int l16  = lane & 15;
  const int m0 = blockIdx.x * 16;
  int arow = m0 + l16; if (arow > N_NODES - 1) arow = N_NODES - 1;
  const float* __restrict__ xr = x + (size_t)arow * DIM_NODE + d.x_off;

  for (int w0 = 0; w0 < d.Wd; w0 += 16) {
    v8f acc = {0.f,0.f,0.f,0.f,0.f,0.f,0.f,0.f};
    const float* __restrict__ wb = W + w0 + l16;
#if HAVE_WMMA_F32
    for (int k0 = 0; k0 < d.U; k0 += 4) {
      const int ka = k0 + 2 * half;
      v2f a, b;
      a.x = xr[(ka + 0) * d.x_stride];
      a.y = xr[(ka + 1) * d.x_stride];
      b.x = wb[(ka + 0) * d.Wd];
      b.y = wb[(ka + 1) * d.Wd];
      acc = __builtin_amdgcn_wmma_f32_16x16x4_f32(false, a, false, b, (short)0, acc, false, false);
    }
#else
    for (int k0 = 0; k0 < d.U; k0 += 32) {
      v16h a, b;
#pragma unroll
      for (int g = 0; g < 8; ++g) {
        const int kb = ((g < 4) ? 2*g : 16 + 2*(g-4)) + half*8 + k0;
        a[2*g]   = (_Float16)xr[(kb + 0) * d.x_stride];
        a[2*g+1] = (_Float16)xr[(kb + 1) * d.x_stride];
        b[2*g]   = (_Float16)wb[(kb + 0) * d.Wd];
        b[2*g+1] = (_Float16)wb[(kb + 1) * d.Wd];
      }
      acc = __builtin_amdgcn_wmma_f32_16x16x32_f16(false, a, false, b, (short)0, acc, false, false);
    }
#endif
    const int col = w0 + l16;
#pragma unroll
    for (int r = 0; r < 8; ++r) {
      const int row = m0 + r + half * 8;
      if (row < N_NODES)
        dst[(size_t)row * d.dst_row_stride + d.dst_off + col * d.dst_stride] = acc[r] * d.scale;
    }
  }
}

// ------------------------------ Edge kernel --------------------------------
// One wave per edge. Gathers the 9.5KB XW record of src (L2-resident),
// contracts with B(i,k)=sum_j e_j C(i,j,k), writes m_ij and atomic-adds into h.
__global__ __launch_bounds__(256) void k_edge(const float* __restrict__ eattr,
                                              const float* __restrict__ alpha,
                                              const int* __restrict__ src,
                                              const int* __restrict__ dstI,
                                              const float* __restrict__ xw,
                                              const float* __restrict__ w3j,
                                              float* __restrict__ h,
                                              float* __restrict__ mij) {
  __shared__ float sw[W3J_TOTAL];
  const int tid = threadIdx.x;
  for (int i = tid; i < W3J_TOTAL; i += 256) sw[i] = w3j[i];

  const int e = blockIdx.x * 8 + (tid >> 5);
  const int lane = tid & 31;
  int sn = 0, dn = 0;
  const bool ok = (e < N_EDGES);
  if (ok) {
    sn = src[e]; dn = dstI[e];
    // warm L2/L1 for the gathered record while we wait on the LDS barrier
    __builtin_prefetch(xw + (size_t)sn * XW_STRIDE + lane * 74, 0, 3);
  }
  __syncthreads();
  if (!ok) return;

  const float* __restrict__ xb = xw + (size_t)sn * XW_STRIDE;
  const float e0 = eattr[e*9 + 0];
  float e1[3], e2[5];
#pragma unroll
  for (int j = 0; j < 3; ++j) e1[j] = eattr[e*9 + 1 + j];
#pragma unroll
  for (int j = 0; j < 5; ++j) e2[j] = eattr[e*9 + 4 + j];
  const float al = alpha[e];

  float o0[4], o1[2][3], o2[5];

  // ---- l3 = 0 : paths (0,0,0), (1,1,0), (2,2,0) ----
  {
    const float b0 = e0 * sw[0];
    float B4[3], B9[5];
#pragma unroll
    for (int i = 0; i < 3; ++i) {
      float s = 0.f;
#pragma unroll
      for (int j = 0; j < 3; ++j) s += e1[j] * sw[44 + i*3 + j];
      B4[i] = s;
    }
#pragma unroll
    for (int i = 0; i < 5; ++i) {
      float s = 0.f;
#pragma unroll
      for (int j = 0; j < 5; ++j) s += e2[j] * sw[213 + i*5 + j];
      B9[i] = s;
    }
#pragma unroll
    for (int t = 0; t < 4; ++t) {
      const int w = lane + 32*t;
      float acc = xb[w] * b0;
#pragma unroll
      for (int i = 0; i < 3; ++i) acc += xb[416 + w*3 + i] * B4[i];
#pragma unroll
      for (int i = 0; i < 5; ++i) acc += xb[1568 + w*5 + i] * B9[i];
      o0[t] = acc;
    }
  }

  // ---- l3 = 1 : paths (0,1,1), (1,0,1), (1,2,1), (2,1,1) ----
  {
    float B1[3], B3[3][3], B6[3][3], B8[5][3];
#pragma unroll
    for (int k = 0; k < 3; ++k) {
      float s = 0.f;
#pragma unroll
      for (int j = 0; j < 3; ++j) s += e1[j] * sw[1 + j*3 + k];
      B1[k] = s;
    }
#pragma unroll
    for (int i = 0; i < 3; ++i)
#pragma unroll
      for (int k = 0; k < 3; ++k) B3[i][k] = e0 * sw[35 + i*3 + k];
#pragma unroll
    for (int i = 0; i < 3; ++i)
#pragma unroll
      for (int k = 0; k < 3; ++k) {
        float s = 0.f;
#pragma unroll
        for (int j = 0; j < 5; ++j) s += e2[j] * sw[98 + (i*5 + j)*3 + k];
        B6[i][k] = s;
      }
#pragma unroll
    for (int i = 0; i < 5; ++i)
#pragma unroll
      for (int k = 0; k < 3; ++k) {
        float s = 0.f;
#pragma unroll
        for (int j = 0; j < 3; ++j) s += e1[j] * sw[168 + (i*3 + j)*3 + k];
        B8[i][k] = s;
      }
#pragma unroll
    for (int t = 0; t < 2; ++t) {
      const int w = lane + 32*t;
      const float xw1 = xb[128 + w];
      float x3[3], x6[3], x8[5];
#pragma unroll
      for (int i = 0; i < 3; ++i) { x3[i] = xb[224 + w*3 + i]; x6[i] = xb[896 + w*3 + i]; }
#pragma unroll
      for (int i = 0; i < 5; ++i) x8[i] = xb[1248 + w*5 + i];
#pragma unroll
      for (int k = 0; k < 3; ++k) {
        float acc = xw1 * B1[k];
#pragma unroll
        for (int i = 0; i < 3; ++i) acc += x3[i]*B3[i][k] + x6[i]*B6[i][k];
#pragma unroll
        for (int i = 0; i < 5; ++i) acc += x8[i]*B8[i][k];
        o1[t][k] = acc;
      }
    }
  }

  // ---- l3 = 2 : paths (0,2,2), (1,1,2), (2,0,2), (2,2,2) ----
  {
    float B2[5], B5[3][5], B7[5][5], B10[5][5];
#pragma unroll
    for (int k = 0; k < 5; ++k) {
      float s = 0.f;
#pragma unroll
      for (int j = 0; j < 5; ++j) s += e2[j] * sw[10 + j*5 + k];
      B2[k] = s;
    }
#pragma unroll
    for (int i = 0; i < 3; ++i)
#pragma unroll
      for (int k = 0; k < 5; ++k) {
        float s = 0.f;
#pragma unroll
        for (int j = 0; j < 3; ++j) s += e1[j] * sw[53 + (i*3 + j)*5 + k];
        B5[i][k] = s;
      }
#pragma unroll
    for (int i = 0; i < 5; ++i)
#pragma unroll
      for (int k = 0; k < 5; ++k) B7[i][k] = e0 * sw[143 + i*5 + k];
#pragma unroll
    for (int i = 0; i < 5; ++i)
#pragma unroll
      for (int k = 0; k < 5; ++k) {
        float s = 0.f;
#pragma unroll
        for (int j = 0; j < 5; ++j) s += e2[j] * sw[238 + (i*5 + j)*5 + k];
        B10[i][k] = s;
      }
    const int w = lane;
    const float xw2 = xb[192 + w];
    float x5[3], x7[5], x10[5];
#pragma unroll
    for (int i = 0; i < 3; ++i) x5[i] = xb[800 + w*3 + i];
#pragma unroll
    for (int i = 0; i < 5; ++i) { x7[i] = xb[1088 + w*5 + i]; x10[i] = xb[2208 + w*5 + i]; }
#pragma unroll
    for (int k = 0; k < 5; ++k) {
      float acc = xw2 * B2[k];
#pragma unroll
      for (int i = 0; i < 3; ++i) acc += x5[i]*B5[i][k];
#pragma unroll
      for (int i = 0; i < 5; ++i) acc += x7[i]*B7[i][k] + x10[i]*B10[i][k];
      o2[k] = acc;
    }
  }

  // ---- write m_ij and scatter-add into h[dst] ----
  float* __restrict__ mo = mij + (size_t)e * DIM_NODE;
  float* __restrict__ hb = h + (size_t)dn * DIM_NODE;
#pragma unroll
  for (int t = 0; t < 4; ++t) {
    const float v = o0[t] * al; const int idx = lane + 32*t;
    mo[idx] = v; atomicAdd(&hb[idx], v);
  }
#pragma unroll
  for (int t = 0; t < 2; ++t)
#pragma unroll
    for (int k = 0; k < 3; ++k) {
      const float v = o1[t][k] * al; const int idx = 128 + (lane + 32*t)*3 + k;
      mo[idx] = v; atomicAdd(&hb[idx], v);
    }
#pragma unroll
  for (int k = 0; k < 5; ++k) {
    const float v = o2[k] * al; const int idx = 320 + lane*5 + k;
    mo[idx] = v; atomicAdd(&hb[idx], v);
  }
}

// --------------------------- Fused node post-pass --------------------------
__device__ inline float sigm(float x) { return 1.f / (1.f + __expf(-x)); }

__global__ __launch_bounds__(256) void k_node(const float* __restrict__ h,
    const float* __restrict__ Wpre0, const float* __restrict__ Wpre1, const float* __restrict__ Wpre2,
    const float* __restrict__ Wpost0, const float* __restrict__ Wpost1, const float* __restrict__ Wpost2,
    const float* __restrict__ ln_g, const float* __restrict__ ln_b,
    const float* __restrict__ mw1, const float* __restrict__ mb1,
    const float* __restrict__ mw2, const float* __restrict__ mb2,
    float* __restrict__ out) {
  extern __shared__ float sm[];
  float* sH = sm;          // 16*480 : h, later gated, later mu/rs/hh scratch
  float* sP = sm + 7680;   // 16*576 : ps, later out(480/node) + gnorm(96/node)
  const int tid = threadIdx.x;
  const int nb = blockIdx.x * 16;

  // 1) stage h
  for (int i = tid; i < 7680; i += 256) {
    const int n = i / 480, gn = nb + n;
    sH[i] = (gn < N_NODES) ? h[(size_t)gn * DIM_NODE + (i % 480)] : 0.f;
  }
  __syncthreads();

  // 2) ps = o3_linear(h, Wpre)
  if (tid < 224) {
    float acc[16];
#pragma unroll
    for (int n = 0; n < 16; ++n) acc[n] = 0.f;
    for (int k = 0; k < 128; ++k) {
      const float wk = Wpre0[k*224 + tid];
#pragma unroll
      for (int n = 0; n < 16; ++n) acc[n] += sH[n*480 + k] * wk;
    }
#pragma unroll
    for (int n = 0; n < 16; ++n) sP[n*576 + tid] = acc[n] * 0.0883883476f;
  }
  if (tid < 192) {
    const int w = tid / 3, i = tid % 3;
    float acc[16];
#pragma unroll
    for (int n = 0; n < 16; ++n) acc[n] = 0.f;
    for (int k = 0; k < 64; ++k) {
      const float wk = Wpre1[k*64 + w];
#pragma unroll
      for (int n = 0; n < 16; ++n) acc[n] += sH[n*480 + 128 + k*3 + i] * wk;
    }
#pragma unroll
    for (int n = 0; n < 16; ++n) sP[n*576 + 224 + tid] = acc[n] * 0.125f;
  }
  if (tid < 160) {
    const int w = tid / 5, i = tid % 5;
    float acc[16];
#pragma unroll
    for (int n = 0; n < 16; ++n) acc[n] = 0.f;
    for (int k = 0; k < 32; ++k) {
      const float wk = Wpre2[k*32 + w];
#pragma unroll
      for (int n = 0; n < 16; ++n) acc[n] += sH[n*480 + 320 + k*5 + i] * wk;
    }
#pragma unroll
    for (int n = 0; n < 16; ++n) sP[n*576 + 416 + tid] = acc[n] * 0.1767766953f;
  }
  __syncthreads();

  // 3) silu / sigmoid gating -> sH
  for (int idx = tid; idx < 7680; idx += 256) {
    const int n = idx / 480, c = idx % 480;
    const float* ps = &sP[n*576];
    float v;
    if (c < 128)      { const float t = ps[c]; v = t * sigm(t); }
    else if (c < 320) { const int cc = c - 128; v = ps[224 + cc] * sigm(ps[128 + cc/3]); }
    else              { const int cc = c - 320; v = ps[416 + cc] * sigm(ps[192 + cc/5]); }
    sH[idx] = v;
  }
  __syncthreads();

  // 4) o3_linear(gated, Wpost) -> sP (stride 480)
  if (tid < 128) {
    float acc[16];
#pragma unroll
    for (int n = 0; n < 16; ++n) acc[n] = 0.f;
    for (int k = 0; k < 128; ++k) {
      const float wk = Wpost0[k*128 + tid];
#pragma unroll
      for (int n = 0; n < 16; ++n) acc[n] += sH[n*480 + k] * wk;
    }
#pragma unroll
    for (int n = 0; n < 16; ++n) sP[n*480 + tid] = acc[n] * 0.0883883476f;
  }
  if (tid < 192) {
    const int w = tid / 3, i = tid % 3;
    float acc[16];
#pragma unroll
    for (int n = 0; n < 16; ++n) acc[n] = 0.f;
    for (int k = 0; k < 64; ++k) {
      const float wk = Wpost1[k*64 + w];
#pragma unroll
      for (int n = 0; n < 16; ++n) acc[n] += sH[n*480 + 128 + k*3 + i] * wk;
    }
#pragma unroll
    for (int n = 0; n < 16; ++n) sP[n*480 + 128 + tid] = acc[n] * 0.125f;
  }
  if (tid < 160) {
    const int w = tid / 5, i = tid % 5;
    float acc[16];
#pragma unroll
    for (int n = 0; n < 16; ++n) acc[n] = 0.f;
    for (int k = 0; k < 32; ++k) {
      const float wk = Wpost2[k*32 + w];
#pragma unroll
      for (int n = 0; n < 16; ++n) acc[n] += sH[n*480 + 320 + k*5 + i] * wk;
    }
#pragma unroll
    for (int n = 0; n < 16; ++n) sP[n*480 + 320 + tid] = acc[n] * 0.1767766953f;
  }
  __syncthreads();

  float* sMu = sH;         // 16
  float* sRs = sH + 16;    // 16
  float* sHH = sH + 64;    // 16*96
  float* sGn = sP + 7680;  // 16*96

  // 5a) LayerNorm stats over the 128 scalars
  if (tid < 16) {
    const int n = tid;
    float s = 0.f, s2 = 0.f;
    for (int c = 0; c < 128; ++c) { const float v = sP[n*480 + c]; s += v; s2 += v*v; }
    const float mu = s * (1.f/128.f);
    const float var = s2 * (1.f/128.f) - mu*mu;
    sMu[n] = mu; sRs[n] = rsqrtf(var + 1e-5f);
  }
  // 5b) vector / quad norms
  for (int idx = tid; idx < 1536; idx += 256) {
    const int n = idx / 96, c = idx % 96;
    float v;
    if (c < 64) {
      const float a0 = sP[n*480 + 128 + c*3 + 0];
      const float a1 = sP[n*480 + 128 + c*3 + 1];
      const float a2 = sP[n*480 + 128 + c*3 + 2];
      v = sqrtf(a0*a0 + a1*a1 + a2*a2);
    } else {
      const int w = c - 64; float s = 0.f;
#pragma unroll
      for (int j = 0; j < 5; ++j) { const float t = sP[n*480 + 320 + w*5 + j]; s += t*t; }
      v = sqrtf(s);
    }
    sGn[idx] = v;
  }
  __syncthreads();

  // 5c) hidden = silu(g @ mlp_w1 + b1)
  for (int idx = tid; idx < 1536; idx += 256) {
    const int n = idx / 96, c = idx % 96;
    float acc = mb1[c];
    for (int k = 0; k < 96; ++k) acc += sGn[n*96 + k] * mw1[k*96 + c];
    sHH[idx] = acc * sigm(acc);
  }
  __syncthreads();

  // 5d) delta + LayerNorm + write scalars
  for (int idx = tid; idx < 2048; idx += 256) {
    const int n = idx / 128, c = idx % 128, gn = nb + n;
    float acc = mb2[c];
    for (int k = 0; k < 96; ++k) acc += sHH[n*96 + k] * mw2[k*128 + c];
    const float s2v = sP[n*480 + c];
    const float so = (s2v - sMu[n]) * sRs[n] * ln_g[c] + ln_b[c] + acc;
    if (gn < N_NODES) out[(size_t)gn * DIM_NODE + c] = so;
  }
  // 5e) pass-through v, q
  for (int idx = tid; idx < 16*352; idx += 256) {
    const int n = idx / 352, c = idx % 352, gn = nb + n;
    if (gn < N_NODES) out[(size_t)gn * DIM_NODE + 128 + c] = sP[n*480 + 128 + c];
  }
}

// --------------------------------- launch ----------------------------------
extern "C" void kernel_launch(void* const* d_in, const int* in_sizes, int n_in,
                              void* d_out, int out_size, void* d_ws, size_t ws_size,
                              hipStream_t stream) {
  (void)in_sizes; (void)n_in; (void)out_size; (void)ws_size;
  const float* x     = (const float*)d_in[0];
  const float* eattr = (const float*)d_in[1];
  const float* alpha = (const float*)d_in[2];
  const int*   src   = (const int*)d_in[3];
  const int*   dst   = (const int*)d_in[4];
  WPtrs wp;
  for (int i = 0; i < 11; ++i) wp.p[i] = (const float*)d_in[5 + i];   // Wtp0..10
  wp.p[11] = (const float*)d_in[16];  // Wself0
  wp.p[12] = (const float*)d_in[17];  // Wself1
  wp.p[13] = (const float*)d_in[18];  // Wself2
  const float* Wpre0  = (const float*)d_in[19];
  const float* Wpre1  = (const float*)d_in[20];
  const float* Wpre2  = (const float*)d_in[21];
  const float* Wpost0 = (const float*)d_in[22];
  const float* Wpost1 = (const float*)d_in[23];
  const float* Wpost2 = (const float*)d_in[24];
  const float* ln_g   = (const float*)d_in[25];
  const float* ln_b   = (const float*)d_in[26];
  const float* mw1    = (const float*)d_in[27];
  const float* mb1    = (const float*)d_in[28];
  const float* mw2    = (const float*)d_in[29];
  const float* mb2    = (const float*)d_in[30];

  float* ws  = (float*)d_ws;
  float* w3j = ws;                                   // 363 (+pad to 512)
  float* h   = ws + 512;                             // 5000*480
  float* xwb = ws + 512 + (size_t)N_NODES*DIM_NODE;  // 5000*2368

  float* out = (float*)d_out;
  float* mij = out + (size_t)N_NODES * DIM_NODE;     // second tuple element

  k_w3j<<<1, 128, 0, stream>>>(w3j);
  k_gemm<<<dim3((N_NODES + 15) / 16, 44), 32, 0, stream>>>(x, wp, xwb, h);
  k_edge<<<N_EDGES / 8, 256, 0, stream>>>(eattr, alpha, src, dst, xwb, w3j, h, mij);
  k_node<<<(N_NODES + 15) / 16, 256, (7680 + 9216) * sizeof(float), stream>>>(
      h, Wpre0, Wpre1, Wpre2, Wpost0, Wpost1, Wpost2,
      ln_g, ln_b, mw1, mb1, mw2, mb2, out);
}
